// TritonAdaptiveLocalConv_30906584662721
// MI455X (gfx1250) — matrix operand order repaired and checked by reading, not compiled, not run on hardware
//
#include <hip/hip_runtime.h>
#include <hip/hip_bf16.h>
#include <math.h>

// ---------------------------------------------------------------------------
// Problem constants (fixed by setup_inputs in the reference)
// ---------------------------------------------------------------------------
#define BB   4
#define LL   2048
#define CC   1024
#define HH   8
#define KK   32
#define DD   (CC / HH)          // 128
#define TOK  (BB * LL)          // 8192
#define HWM  16                 // max_window = min(int(sqrt(2048))=45, 32)=32 ; hwm=16
#define MAXW 32.0f
#define MINW 4.0f
#define MAXOFF 45.0f            // float(int(sqrt(2048)))
#define EPS 1e-6f

typedef __attribute__((ext_vector_type(16))) __bf16       v16bf;
typedef __attribute__((ext_vector_type(8)))  float        v8f;
typedef __attribute__((ext_vector_type(4)))  float        vf4;
typedef __attribute__((ext_vector_type(4)))  unsigned int v4u;
typedef __attribute__((ext_vector_type(4)))  int          v4i;
typedef __attribute__((ext_vector_type(2)))  unsigned int v2u;

#if defined(__has_builtin)
#  if __has_builtin(__builtin_amdgcn_global_load_async_to_lds_b128)
#    define HAVE_ASYNC_B128 1
#  endif
#  if __has_builtin(__builtin_amdgcn_s_wait_asynccnt)
#    define HAVE_WAIT_ASYNC 1
#  endif
#endif

static __device__ __forceinline__ unsigned short f2bf(float f) {
    union { __bf16 h; unsigned short u; } c;
    c.h = (__bf16)f;           // RNE convert
    return c.u;
}
static __device__ __forceinline__ float silu(float x) {
    return x / (1.0f + __expf(-x));
}

// 16-byte global -> LDS copy; async (ASYNCcnt) path when toolchain exposes it.
static __device__ __forceinline__ void copy16_g2l(const __bf16* g, __bf16* l) {
#if defined(HAVE_ASYNC_B128)
    __builtin_amdgcn_global_load_async_to_lds_b128((v4i*)(g), (v4i*)(l), 0, 0);
#else
    *(v4u*)l = *(const v4u*)g;
#endif
}
static __device__ __forceinline__ void async_wait_all() {
#if defined(HAVE_ASYNC_B128)
#  if defined(HAVE_WAIT_ASYNC)
    __builtin_amdgcn_s_wait_asynccnt(0);
#  else
    asm volatile("s_wait_asynccnt 0x0" ::: "memory");
#  endif
#endif
}

// ---------------------------------------------------------------------------
// Kernel 0: bulk f32 -> bf16 conversion
// ---------------------------------------------------------------------------
__global__ __launch_bounds__(256) void f32_to_bf16_kernel(
    const float* __restrict__ in, __bf16* __restrict__ out, int n)
{
    const int i = (blockIdx.x * 256 + threadIdx.x) * 4;
    if (i + 3 < n) {
        vf4 v = *(const vf4*)(in + i);
        v2u p;
        p[0] = ((unsigned)f2bf(v[1]) << 16) | f2bf(v[0]);
        p[1] = ((unsigned)f2bf(v[3]) << 16) | f2bf(v[2]);
        *(v2u*)(out + i) = p;
    }
}

// ---------------------------------------------------------------------------
// Kernel 1: window & offset projections (N=8 each) + RMSNorm + sigmoid/tanh.
// One wave per token; butterfly reductions leave full sums in every lane.
// ---------------------------------------------------------------------------
__global__ __launch_bounds__(256) void winoff_kernel(
    const float* __restrict__ x,
    const float* __restrict__ ww, const float* __restrict__ wb, const float* __restrict__ wg,
    const float* __restrict__ ow, const float* __restrict__ ob, const float* __restrict__ og,
    float* __restrict__ hw_out, float* __restrict__ co_out)
{
    const int lane = threadIdx.x & 31;
    const int t = blockIdx.x * 8 + (threadIdx.x >> 5);

    float accW[HH], accO[HH];
#pragma unroll
    for (int h = 0; h < HH; ++h) { accW[h] = 0.f; accO[h] = 0.f; }

    const float* xr = x + (size_t)t * CC;
    for (int k = lane; k < CC; k += 32) {
        float xv = xr[k];
#pragma unroll
        for (int h = 0; h < HH; ++h) {
            accW[h] = fmaf(xv, ww[h * CC + k], accW[h]);
            accO[h] = fmaf(xv, ow[h * CC + k], accO[h]);
        }
    }
#pragma unroll
    for (int h = 0; h < HH; ++h) {
#pragma unroll
        for (int s = 16; s > 0; s >>= 1) {
            accW[h] += __shfl_xor(accW[h], s, 32);
            accO[h] += __shfl_xor(accO[h], s, 32);
        }
    }
    float wr[HH], orr[HH];
    float msw = 0.f, mso = 0.f;
#pragma unroll
    for (int h = 0; h < HH; ++h) {
        wr[h]  = accW[h] + wb[h];
        orr[h] = accO[h] + ob[h];
        msw += wr[h] * wr[h];
        mso += orr[h] * orr[h];
    }
    const float rw = sqrtf(msw / (float)HH);
    const float ro = sqrtf(mso / (float)HH);
    if (lane < HH) {
        float wn = wr[lane] * wg[lane] / (rw + EPS);
        float sg = 1.0f / (1.0f + __expf(-wn));
        float wsz = MINW + sg * (MAXW - MINW);
        hw_out[t * HH + lane] = 0.5f * wsz;

        float on = orr[lane] * og[lane] / (ro + EPS);
        co_out[t * HH + lane] = tanhf(on) * MAXOFF;
    }
}

// ---------------------------------------------------------------------------
// Kernel 2: tiled bf16 WMMA GEMM  out[M,N] = A[M,K] @ W[N,K]^T (+bias)(+SiLU)
// A,W already bf16. 128x128 block, BK=64, DOUBLE-BUFFERED async global->LDS:
// tile k+1 streams in (ASYNCcnt) while 16 WMMAs run on tile k; one barrier
// per K-step. 8 waves = 4(M) x 2(N); each wave: 2x4 f32 16x16 accumulators.
// ---------------------------------------------------------------------------
#define BKH  64   // K halfs staged per iteration
#define LDP2 72   // padded LDS row in halfs: 144 B, 16B-aligned, conflict-friendly

__global__ __launch_bounds__(256) void wmma_gemm_bf16(
    const __bf16* __restrict__ A, const __bf16* __restrict__ W,
    const float* __restrict__ bias, float* __restrict__ out,
    int M, int N, int Kdim, int act)
{
    __shared__ __bf16 As[2][128][LDP2];
    __shared__ __bf16 Bs[2][128][LDP2];

    const int tid  = threadIdx.x;
    const int lane = tid & 31;
    const int wid  = tid >> 5;
    const int wm   = wid & 3;   // M offset wm*32
    const int wn   = wid >> 2;  // N offset wn*64
    const int m0   = blockIdx.y * 128;
    const int n0   = blockIdx.x * 128;

    // staging: each thread copies 32 halfs (64B) of each tile
    const int srow = tid >> 1;          // 0..127
    const int scol = (tid & 1) * 32;    // 0 or 32 (halfs)

    const __bf16* gA = A + (size_t)(m0 + srow) * Kdim + scol;
    const __bf16* gB = W + (size_t)(n0 + srow) * Kdim + scol;

    auto stage = [&](int buf, int kk) {
        const __bf16* pa = gA + kk;
        const __bf16* pb = gB + kk;
        if (kk + BKH < Kdim) {          // prefetch the tile after next
            __builtin_prefetch(pa + BKH, 0, 3);
            __builtin_prefetch(pb + BKH, 0, 3);
        }
#pragma unroll
        for (int j = 0; j < 32; j += 8) {
            copy16_g2l(pa + j, &As[buf][srow][scol + j]);
            copy16_g2l(pb + j, &Bs[buf][srow][scol + j]);
        }
    };

    v8f acc[2][4] = {};
    union Frag { v4u q[2]; v16bf m; };

    const int lm  = lane & 15;
    const int ab0 = (lane >> 4) * 8;    // A frag K sub-base (halfs)
    const int bk0 = (lane >> 4) * 16;   // unused; B frags use ab0 like A

    stage(0, 0);                        // prologue: tile 0 in flight
    int cur = 0;

    for (int k0 = 0; k0 < Kdim; k0 += BKH) {
        async_wait_all();               // this wave's loads into `cur` landed
        __syncthreads();                // everyone's landed; `next` free to reuse
        if (k0 + BKH < Kdim)
            stage(cur ^ 1, k0 + BKH);   // overlap next tile's DMA with WMMAs

#pragma unroll
        for (int ks = 0; ks < BKH; ks += 32) {
            Frag afr[2], bfr[4];
#pragma unroll
            for (int mt = 0; mt < 2; ++mt) {
                const int r = wm * 32 + mt * 16 + lm;
                afr[mt].q[0] = *(const v4u*)&As[cur][r][ks + ab0];
                afr[mt].q[1] = *(const v4u*)&As[cur][r][ks + 16 + ab0];
            }
#pragma unroll
            for (int nt = 0; nt < 4; ++nt) {
                const int r = wn * 64 + nt * 16 + lm;
                bfr[nt].q[0] = *(const v4u*)&Bs[cur][r][ks + ab0];
                bfr[nt].q[1] = *(const v4u*)&Bs[cur][r][ks + 16 + ab0];
            }
#pragma unroll
            for (int mt = 0; mt < 2; ++mt)
#pragma unroll
                for (int nt = 0; nt < 4; ++nt)
                    acc[mt][nt] = __builtin_amdgcn_wmma_f32_16x16x32_bf16(
                        false, afr[mt].m, false, bfr[nt].m,
                        (short)0, acc[mt][nt], false, false);
        }
        cur ^= 1;
    }

    // epilogue: C/D layout -> VGPR r: lanes0-15 M=r, lanes16-31 M=8+r
    const int mhi = (lane >> 4) * 8;
#pragma unroll
    for (int mt = 0; mt < 2; ++mt) {
#pragma unroll
        for (int nt = 0; nt < 4; ++nt) {
            const int col = n0 + wn * 64 + nt * 16 + lm;
            const float bv = bias ? bias[col] : 0.0f;
#pragma unroll
            for (int r = 0; r < 8; ++r) {
                const int row = m0 + wm * 32 + mt * 16 + mhi + r;
                float v = acc[mt][nt][r] + bv;
                if (act == 1) v = silu(v);
                out[(size_t)row * N + col] = v;
            }
        }
    }
}

// ---------------------------------------------------------------------------
// Kernel 3: per-token RMSNorm + SiLU over the 256 kernel logits (in place)
// ---------------------------------------------------------------------------
__global__ __launch_bounds__(256) void knorm_kernel(
    float* __restrict__ kact, const float* __restrict__ kg)
{
    const int lane = threadIdx.x & 31;
    const int t = blockIdx.x * 8 + (threadIdx.x >> 5);
    const int NK = HH * KK;              // 256

    float v[8];
    float ss = 0.f;
#pragma unroll
    for (int i = 0; i < 8; ++i) {
        v[i] = kact[(size_t)t * NK + i * 32 + lane];
        ss = fmaf(v[i], v[i], ss);
    }
#pragma unroll
    for (int s = 16; s > 0; s >>= 1) ss += __shfl_xor(ss, s, 32);
    const float r = sqrtf(ss / (float)NK);
#pragma unroll
    for (int i = 0; i < 8; ++i) {
        const int c = i * 32 + lane;
        float nv = v[i] * kg[c] / (r + EPS);
        kact[(size_t)t * NK + c] = silu(nv);
    }
}

// ---------------------------------------------------------------------------
// Kernel 4: adaptive windowed gather-conv. One wave per (b,l,h); lane owns 4 D.
// Window/offset scalars are wave-uniform -> tap skipping is divergence-free.
// Emits hidden directly in bf16 (feeds the output WMMA GEMM).
// ---------------------------------------------------------------------------
__global__ __launch_bounds__(256) void adaptconv_kernel(
    const float* __restrict__ v4, const float* __restrict__ hw,
    const float* __restrict__ co, const float* __restrict__ kact,
    __bf16* __restrict__ hidden)
{
    const int lane = threadIdx.x & 31;
    const int g = blockIdx.x * 8 + (threadIdx.x >> 5);   // 0 .. TOK*HH-1
    const int h = g % HH;
    const int tl = g / HH;          // token index b*L + l
    const int l = tl % LL;
    const int b = tl / LL;

    const int s = tl * HH + h;
    const float hwv = hw[s];
    const float cov = co[s];
    const float* kr = kact + (size_t)s * KK;
    const int d0 = lane * 4;

    vf4 acc = {0.f, 0.f, 0.f, 0.f};

#pragma unroll
    for (int j = -HWM; j <= HWM; ++j) {
        const float jf = (float)j;
        float m = hwv + 0.5f - fabsf(jf);
        m = fminf(fmaxf(m, 0.f), 1.f);
        if (m <= 0.f) continue;                 // wave-uniform skip
        int kidx = j + HWM; if (kidx > KK - 1) kidx = KK - 1;  // jnp.take clip
        const float w = m * kr[kidx];

        const float pos = (float)l + cov + jf;
        const float p0f = floorf(pos);
        const float fr = pos - p0f;
        const int p0 = (int)p0f;

        vf4 g0 = {0.f, 0.f, 0.f, 0.f}, g1 = {0.f, 0.f, 0.f, 0.f};
        if (p0 >= 0 && p0 < LL)
            g0 = *(const vf4*)(v4 + (((size_t)(b * LL + p0) * HH + h) * DD) + d0);
        const int p1 = p0 + 1;
        if (p1 >= 0 && p1 < LL)
            g1 = *(const vf4*)(v4 + (((size_t)(b * LL + p1) * HH + h) * DD) + d0);

        acc = acc + g0 * (w * (1.0f - fr)) + g1 * (w * fr);
    }
    v2u pk;
    pk[0] = ((unsigned)f2bf(acc[1]) << 16) | f2bf(acc[0]);
    pk[1] = ((unsigned)f2bf(acc[3]) << 16) | f2bf(acc[2]);
    *(v2u*)(hidden + (size_t)tl * CC + h * DD + d0) = pk;
}

// ---------------------------------------------------------------------------
// Host launcher
// ---------------------------------------------------------------------------
extern "C" void kernel_launch(void* const* d_in, const int* in_sizes, int n_in,
                              void* d_out, int out_size, void* d_ws, size_t ws_size,
                              hipStream_t stream) {
    const float* x   = (const float*)d_in[0];
    const float* ww  = (const float*)d_in[1];
    const float* wb  = (const float*)d_in[2];
    const float* wg  = (const float*)d_in[3];
    const float* ow  = (const float*)d_in[4];
    const float* ob  = (const float*)d_in[5];
    const float* og  = (const float*)d_in[6];
    const float* kw  = (const float*)d_in[7];
    const float* kb  = (const float*)d_in[8];
    const float* kg  = (const float*)d_in[9];
    const float* vw  = (const float*)d_in[10];
    const float* vb  = (const float*)d_in[11];
    const float* outw= (const float*)d_in[12];
    float* out = (float*)d_out;

    // workspace partition
    char* base = (char*)d_ws;
    size_t off = 0;
    float*  hwbuf  = (float*)(base + off);  off += (size_t)TOK * HH * sizeof(float);
    float*  cobuf  = (float*)(base + off);  off += (size_t)TOK * HH * sizeof(float);
    float*  kact   = (float*)(base + off);  off += (size_t)TOK * HH * KK * sizeof(float);
    float*  v4buf  = (float*)(base + off);  off += (size_t)TOK * CC * sizeof(float);
    __bf16* hidbf  = (__bf16*)(base + off); off += (size_t)TOK * CC * sizeof(__bf16);
    __bf16* xbf    = (__bf16*)(base + off); off += (size_t)TOK * CC * sizeof(__bf16);
    __bf16* kwbf   = (__bf16*)(base + off); off += (size_t)HH * KK * CC * sizeof(__bf16);
    __bf16* vwbf   = (__bf16*)(base + off); off += (size_t)CC * CC * sizeof(__bf16);
    __bf16* owbf   = (__bf16*)(base + off); off += (size_t)CC * CC * sizeof(__bf16);
    (void)off; (void)ws_size; (void)n_in; (void)in_sizes; (void)out_size;

    // 0) one-time f32 -> bf16 conversions (x feeds 2 GEMMs, weights 1 each)
    {
        int nx = TOK * CC;            // 8388608
        f32_to_bf16_kernel<<<(nx / 4 + 255) / 256, 256, 0, stream>>>(x, xbf, nx);
        int nk = HH * KK * CC;        // 262144
        f32_to_bf16_kernel<<<(nk / 4 + 255) / 256, 256, 0, stream>>>(kw, kwbf, nk);
        int nv = CC * CC;             // 1048576
        f32_to_bf16_kernel<<<(nv / 4 + 255) / 256, 256, 0, stream>>>(vw, vwbf, nv);
        f32_to_bf16_kernel<<<(nv / 4 + 255) / 256, 256, 0, stream>>>(outw, owbf, nv);
    }

    // 1) window/offset projections + norms
    winoff_kernel<<<TOK / 8, 256, 0, stream>>>(x, ww, wb, wg, ow, ob, og, hwbuf, cobuf);

    // 2) kernel projection GEMM (N=256) -> raw logits into kact
    wmma_gemm_bf16<<<dim3((HH * KK) / 128, TOK / 128), 256, 0, stream>>>(
        xbf, kwbf, kb, kact, TOK, HH * KK, CC, 0);

    // 3) RMSNorm + SiLU in place on kact
    knorm_kernel<<<TOK / 8, 256, 0, stream>>>(kact, kg);

    // 4) v projection GEMM (N=1024) -> f32 v4 (conv interpolates in f32)
    wmma_gemm_bf16<<<dim3(CC / 128, TOK / 128), 256, 0, stream>>>(
        xbf, vwbf, vb, v4buf, TOK, CC, CC, 0);

    // 5) adaptive conv -> hidden (bf16)
    adaptconv_kernel<<<(TOK * HH) / 8, 256, 0, stream>>>(v4buf, hwbuf, cobuf, kact, hidbf);

    // 6) output GEMM + SiLU -> d_out
    wmma_gemm_bf16<<<dim3(CC / 128, TOK / 128), 256, 0, stream>>>(
        hidbf, owbf, nullptr, out, TOK, CC, CC, 1);
}
